// ParallelGRU_30872224923912
// MI455X (gfx1250) — compile-verified, matrix operand order checked by reading, not run on hardware
//
#include <hip/hip_runtime.h>
#include <hip/hip_bf16.h>

typedef __attribute__((ext_vector_type(16))) __bf16 v16bf;
typedef __attribute__((ext_vector_type(8)))  float  v8f;

#define An 8
#define Bn 64
#define Sn 128
#define En 512
#define Hn 512

union AFrag { v16bf v; uint4 q[2]; };

__device__ __forceinline__ unsigned int f2bf(float f) {
    unsigned int u = __float_as_uint(f);
    // round-to-nearest-even bf16
    unsigned int r = (u + 0x7FFFu + ((u >> 16) & 1u)) >> 16;
    return r & 0xFFFFu;
}

__device__ __forceinline__ void wait_async_lds() {
    asm volatile("s_wait_asynccnt 0x0" ::: "memory");
}

// ---------------------------------------------------------------------------
// Kernel 1: pack the six fp32 weight matrices (A,512,512) into bf16 WMMA
// B-fragment order. frag(w,a,kb,ntg) = 32 lanes x 16 bf16; lane holds column
// n = ntg*16 + lane%16, K pattern: K = (i&7)|(khi<<3)|((i&8)<<1), khi=lane/16.
// ---------------------------------------------------------------------------
__global__ __launch_bounds__(256) void gru_pack_weights(
    const float* __restrict__ i2r, const float* __restrict__ i2z,
    const float* __restrict__ i2n, const float* __restrict__ h2r,
    const float* __restrict__ h2z, const float* __restrict__ h2n,
    unsigned int* __restrict__ packed)
{
    int t    = blockIdx.x * 256 + threadIdx.x;     // 786432 total
    int lane = t & 31;
    int frag = t >> 5;
    int ntg  = frag & 31;
    int kb   = (frag >> 5) & 15;
    int a    = (frag >> 9) & 7;
    int w    = frag >> 12;                         // 0..5
    const float* W = (w == 0) ? i2r : (w == 1) ? i2z : (w == 2) ? i2n
                   : (w == 3) ? h2r : (w == 4) ? h2z : h2n;
    int n   = ntg * 16 + (lane & 15);
    int khi = lane >> 4;

    unsigned int outw[8];
#pragma unroll
    for (int j = 0; j < 8; ++j) {
        int i0 = 2 * j;
        int K0 = (i0 & 7) | (khi << 3) | ((i0 & 8) << 1);
        unsigned int lo = f2bf(W[(a * En + kb * 32 + K0) * Hn + n]);
        unsigned int hi = f2bf(W[(a * En + kb * 32 + K0 + 1) * Hn + n]);
        outw[j] = lo | (hi << 16);
    }
    unsigned int* dst = packed +
        ((((unsigned)(w * An + a) * 16 + kb) * 32 + ntg) * 32 + lane) * 8;
    *(uint4*)(dst)     = make_uint4(outw[0], outw[1], outw[2], outw[3]);
    *(uint4*)(dst + 4) = make_uint4(outw[4], outw[5], outw[6], outw[7]);
}

// ---------------------------------------------------------------------------
// Kernel 2: pack x (A,B,S,E fp32) into bf16 A-fragment order, one contiguous
// 64 KB block per (a,s): [(a*S+s)][mt][kb][lane][16 bf16]. Done once; each
// step then async-copies its block straight into LDS.
// ---------------------------------------------------------------------------
__global__ __launch_bounds__(256) void gru_pack_x(
    const float* __restrict__ x, unsigned int* __restrict__ xfrag)
{
    int t    = blockIdx.x * 256 + threadIdx.x;     // A*S*4*16*32 = 2097152
    int lane = t & 31;
    int frag = t >> 5;                             // ((a*128+s)*4+mt)*16+kb
    int kb   = frag & 15;
    int mt   = (frag >> 4) & 3;
    int s    = (frag >> 6) & 127;
    int a    = frag >> 13;
    int b    = mt * 16 + (lane & 15);
    int khi  = lane >> 4;
    const float* src = x + ((size_t)(a * Bn + b) * Sn + s) * En + kb * 32;

    unsigned int outw[8];
#pragma unroll
    for (int j = 0; j < 8; ++j) {
        int i0 = 2 * j;
        int K0 = (i0 & 7) | (khi << 3) | ((i0 & 8) << 1);
        unsigned int lo = f2bf(src[K0]);
        unsigned int hi = f2bf(src[K0 + 1]);
        outw[j] = lo | (hi << 16);
    }
    unsigned int* dst = xfrag + ((size_t)frag * 32 + lane) * 8;
    *(uint4*)(dst)     = make_uint4(outw[0], outw[1], outw[2], outw[3]);
    *(uint4*)(dst + 4) = make_uint4(outw[4], outw[5], outw[6], outw[7]);
}

// ---------------------------------------------------------------------------
// Kernel 3: one GRU time step. Grid = A * (H/64) = 64 blocks, 256 threads.
// Waves 0-3: input-side GEMMs, waves 4-7: hidden-side GEMMs; each wave owns a
// 16-row M tile and 64 output columns. x A-fragments arrive via
// global_load_async_to_lds_b128 (ASYNCcnt), h_prev is converted fp32->bf16
// into fragment order while those fly. B-fragments stream from L2-resident
// packed weights via global_load_b128.
// ---------------------------------------------------------------------------
__global__ __launch_bounds__(256) void gru_step(
    const unsigned int* __restrict__ xfrag,
    const float* __restrict__ bir, const float* __restrict__ biz,
    const float* __restrict__ bin, const float* __restrict__ bhr,
    const float* __restrict__ bhz, const float* __restrict__ bhn,
    const unsigned int* __restrict__ packedW,
    float* __restrict__ out, float* __restrict__ hlast, int s)
{
    extern __shared__ unsigned int sm[];   // 32768 dwords = 128 KB dynamic LDS
    // [0, 16384)     : x A-fragments (64 KB), reused as hidSum (48 KB) later
    // [16384, 32768) : h A-fragments (64 KB)
    const int a    = blockIdx.x >> 3;
    const int nb   = blockIdx.x & 7;
    const int tid  = threadIdx.x;
    const int lane = tid & 31;
    const int wave = tid >> 5;

    // ---- async-copy this (a,s) x-fragment block (64 KB contiguous) to LDS.
    // GV mode: the offset is added to both the LDS and global address; the dynamic LDS
    // allocation starts at wave LDS base, so raw byte offsets are correct.
    {
        const unsigned int* xsrc = xfrag + (size_t)(a * Sn + s) * 16384u;
        unsigned int lds_addr = (unsigned int)tid * 16u;
        unsigned long long ga =
            (unsigned long long)(uintptr_t)xsrc + (unsigned long long)tid * 16ull;
        asm volatile("global_load_async_to_lds_b128 %0, %1, off offset:0"    :: "v"(lds_addr), "v"(ga) : "memory");
        asm volatile("global_load_async_to_lds_b128 %0, %1, off offset:4096" :: "v"(lds_addr), "v"(ga) : "memory");
        asm volatile("global_load_async_to_lds_b128 %0, %1, off offset:8192" :: "v"(lds_addr), "v"(ga) : "memory");
        asm volatile("global_load_async_to_lds_b128 %0, %1, off offset:12288":: "v"(lds_addr), "v"(ga) : "memory");
        asm volatile("global_load_async_to_lds_b128 %0, %1, off offset:16384":: "v"(lds_addr), "v"(ga) : "memory");
        asm volatile("global_load_async_to_lds_b128 %0, %1, off offset:20480":: "v"(lds_addr), "v"(ga) : "memory");
        asm volatile("global_load_async_to_lds_b128 %0, %1, off offset:24576":: "v"(lds_addr), "v"(ga) : "memory");
        asm volatile("global_load_async_to_lds_b128 %0, %1, off offset:28672":: "v"(lds_addr), "v"(ga) : "memory");
        asm volatile("global_load_async_to_lds_b128 %0, %1, off offset:32768":: "v"(lds_addr), "v"(ga) : "memory");
        asm volatile("global_load_async_to_lds_b128 %0, %1, off offset:36864":: "v"(lds_addr), "v"(ga) : "memory");
        asm volatile("global_load_async_to_lds_b128 %0, %1, off offset:40960":: "v"(lds_addr), "v"(ga) : "memory");
        asm volatile("global_load_async_to_lds_b128 %0, %1, off offset:45056":: "v"(lds_addr), "v"(ga) : "memory");
        asm volatile("global_load_async_to_lds_b128 %0, %1, off offset:49152":: "v"(lds_addr), "v"(ga) : "memory");
        asm volatile("global_load_async_to_lds_b128 %0, %1, off offset:53248":: "v"(lds_addr), "v"(ga) : "memory");
        asm volatile("global_load_async_to_lds_b128 %0, %1, off offset:57344":: "v"(lds_addr), "v"(ga) : "memory");
        asm volatile("global_load_async_to_lds_b128 %0, %1, off offset:61440":: "v"(lds_addr), "v"(ga) : "memory");
    }

    // ---- stage h_prev as bf16 A-fragments (overlaps with async x copy) ----
    {
        int b  = tid >> 2;                 // row 0..63
        int e0 = (tid & 3) * 128;
        const float* hrow = (s == 0) ? nullptr
                          : out + ((size_t)(a * Bn + b) * Sn + (s - 1)) * Hn;
        int mt = b >> 4, M = b & 15;
        for (int e = e0; e < e0 + 128; e += 2) {
            int kb = e >> 5, Klo = e & 31;
            int laneT = M + (((Klo >> 3) & 1) << 4);
            int idx   = (Klo & 7) | ((Klo >> 4) << 3);
            unsigned int off = ((unsigned)(mt * 16 + kb) * 32 + laneT) * 8u + (idx >> 1);
            unsigned int hv = 0;
            if (hrow) { float2 h2 = *(const float2*)(hrow + e);
                        hv = f2bf(h2.x) | (f2bf(h2.y) << 16); }
            sm[16384 + off] = hv;
        }
    }
    wait_async_lds();
    __syncthreads();

    const int side = wave >> 2;            // 0: input-side, 1: hidden-side
    const int mt   = wave & 3;
    v8f acc[12] = {};
    const unsigned int aBase = side ? 16384u : 0u;

    for (int kb = 0; kb < 16; ++kb) {
        AFrag af;
        const uint4* q = (const uint4*)&sm[aBase + ((unsigned)(mt * 16 + kb) * 32 + lane) * 8u];
        af.q[0] = q[0]; af.q[1] = q[1];
#pragma unroll
        for (int g = 0; g < 3; ++g) {
            int w = side * 3 + g;
#pragma unroll
            for (int nt = 0; nt < 4; ++nt) {
                int ntg = nb * 4 + nt;
                AFrag bf_;
                const uint4* p = (const uint4*)(packedW +
                    ((((unsigned)(w * An + a) * 16 + kb) * 32 + ntg) * 32 + lane) * 8);
                bf_.q[0] = p[0]; bf_.q[1] = p[1];
                acc[g * 4 + nt] = __builtin_amdgcn_wmma_f32_16x16x32_bf16(
                    false, af.v, false, bf_.v, (short)0, acc[g * 4 + nt], false, false);
            }
        }
    }
    __syncthreads();

    float* hid = (float*)sm;               // hidSum[3][64][64] reuses x region
    if (side == 1) {
#pragma unroll
        for (int g = 0; g < 3; ++g)
#pragma unroll
            for (int nt = 0; nt < 4; ++nt)
#pragma unroll
                for (int j = 0; j < 8; ++j) {
                    int m = mt * 16 + j + ((lane >> 4) << 3);
                    int n = (lane & 15) + nt * 16;
                    hid[(g * 64 + m) * 64 + n] = acc[g * 4 + nt][j];
                }
    }
    __syncthreads();

    if (side == 0) {
#pragma unroll
        for (int nt = 0; nt < 4; ++nt) {
            int n    = (lane & 15) + nt * 16;
            int ncol = nb * 64 + n;
            float vbir = bir[a * Hn + ncol], vbiz = biz[a * Hn + ncol];
            float vbin = bin[a * Hn + ncol], vbhr = bhr[a * Hn + ncol];
            float vbhz = bhz[a * Hn + ncol], vbhn = bhn[a * Hn + ncol];
#pragma unroll
            for (int j = 0; j < 8; ++j) {
                int m = mt * 16 + j + ((lane >> 4) << 3);
                float xr = acc[0 * 4 + nt][j] + vbir;
                float xz = acc[1 * 4 + nt][j] + vbiz;
                float xn = acc[2 * 4 + nt][j] + vbin;
                float hr = hid[(0 * 64 + m) * 64 + n] + vbhr;
                float hz = hid[(1 * 64 + m) * 64 + n] + vbhz;
                float hn = hid[(2 * 64 + m) * 64 + n] + vbhn;
                float r  = 1.0f / (1.0f + __expf(-(xr + hr)));
                float z  = 1.0f / (1.0f + __expf(-(xz + hz)));
                float nv = tanhf(xn + r * hn);
                float hp = (s == 0) ? 0.0f
                         : out[((size_t)(a * Bn + m) * Sn + (s - 1)) * Hn + ncol];
                float hnew = (1.0f - z) * nv + z * hp;
                out[((size_t)(a * Bn + m) * Sn + s) * Hn + ncol] = hnew;
                if (s == Sn - 1)
                    hlast[(size_t)(a * Bn + m) * Hn + ncol] = hnew;
            }
        }
    }
}

extern "C" void kernel_launch(void* const* d_in, const int* in_sizes, int n_in,
                              void* d_out, int out_size, void* d_ws, size_t ws_size,
                              hipStream_t stream) {
    (void)in_sizes; (void)n_in; (void)out_size; (void)ws_size;
    const float* x   = (const float*)d_in[0];
    const float* i2r = (const float*)d_in[1];
    const float* i2z = (const float*)d_in[2];
    const float* i2n = (const float*)d_in[3];
    const float* h2r = (const float*)d_in[4];
    const float* h2z = (const float*)d_in[5];
    const float* h2n = (const float*)d_in[6];
    const float* bir = (const float*)d_in[7];
    const float* biz = (const float*)d_in[8];
    const float* bin = (const float*)d_in[9];
    const float* bhr = (const float*)d_in[10];
    const float* bhz = (const float*)d_in[11];
    const float* bhn = (const float*)d_in[12];

    unsigned int* packed = (unsigned int*)d_ws;          // 24 MB B-fragments
    unsigned int* xfrag  = packed + (size_t)6 * An * 16 * 32 * 32 * 8; // +64 MB
    float* out   = (float*)d_out;
    float* hlast = out + (size_t)An * Bn * Sn * Hn;

    gru_pack_weights<<<3072, 256, 0, stream>>>(i2r, i2z, i2n, h2r, h2z, h2n, packed);
    gru_pack_x<<<8192, 256, 0, stream>>>(x, xfrag);

    for (int s = 0; s < Sn; ++s) {
        gru_step<<<An * (Hn / 64), 256, 128 * 1024, stream>>>(
            xfrag, bir, biz, bin, bhr, bhz, bhn, packed, out, hlast, s);
    }
}
